// Euclid_FC_60078002536974
// MI455X (gfx1250) — matrix operand order repaired and checked by reading, not compiled, not run on hardware
//
#include <hip/hip_runtime.h>

typedef __attribute__((ext_vector_type(16))) __bf16 v16bf;
typedef __attribute__((ext_vector_type(4)))  __bf16 v4bf;
typedef __attribute__((ext_vector_type(8)))  float  v8f;
typedef __attribute__((ext_vector_type(4)))  float  v4f;
typedef __attribute__((ext_vector_type(4)))  unsigned int v4u;
typedef __attribute__((ext_vector_type(8)))  unsigned int v8u;
typedef __attribute__((ext_vector_type(4)))  int v4i;

#define BATCH 2048
#define DIN   1024
#define DOUT  4096

#if __has_builtin(__builtin_amdgcn_global_load_async_to_lds_b128) && \
    __has_builtin(__builtin_amdgcn_s_wait_asynccnt)
#define ASYNC_OK 1
#else
#define ASYNC_OK 0
#endif

// ---------------------------------------------------------------------------
// helpers
// ---------------------------------------------------------------------------
__device__ __forceinline__ void splitf(float f, int i, v16bf& hi, v16bf& lo) {
  __bf16 h = (__bf16)f;
  hi[i] = h;
  lo[i] = (__bf16)(f - (float)h);
}

struct FragBits { v4u a, b; };

// load a v16bf fragment as two 16B chunks at p and p+gapElems
__device__ __forceinline__ v16bf ldfrag(const __bf16* p, int gapElems) {
  FragBits u;
  u.a = *(const v4u*)p;
  u.b = *(const v4u*)(p + gapElems);
  return __builtin_bit_cast(v16bf, u);
}

// 16B global -> LDS copy: async on CDNA5 (ASYNCcnt), sync fallback otherwise
__device__ __forceinline__ void cp16_g2l(const __bf16* g, __bf16* l) {
#if ASYNC_OK
  __builtin_amdgcn_global_load_async_to_lds_b128(
      (__attribute__((address_space(1))) v4i*)(void*)g,
      (__attribute__((address_space(3))) v4i*)(void*)l, 0, 0);
#else
  *(v4u*)l = *(const v4u*)g;
#endif
}

__device__ __forceinline__ void wait_async0() {
#if ASYNC_OK
  __builtin_amdgcn_s_wait_asynccnt(0);
#endif
}

// ---------------------------------------------------------------------------
// pack x -> Xhi/Xlo (row-major bf16) + x_sq   (one wave per row)
// ---------------------------------------------------------------------------
__global__ __launch_bounds__(256) void pack_x_kernel(const float* __restrict__ x,
                                                     __bf16* __restrict__ Xhi,
                                                     __bf16* __restrict__ Xlo,
                                                     float* __restrict__ xs) {
  int row  = blockIdx.x * 8 + (threadIdx.x >> 5);
  int lane = threadIdx.x & 31;
  const float* p = x + (size_t)row * DIN;
  __bf16* ph = Xhi + (size_t)row * DIN;
  __bf16* pl = Xlo + (size_t)row * DIN;
  float s = 0.f;
  #pragma unroll
  for (int c = 0; c < 8; ++c) {
    int i = c * 128 + lane * 4;
    v4f v = *(const v4f*)(p + i);
    v4bf h, l;
    #pragma unroll
    for (int t = 0; t < 4; ++t) {
      float f = v[t];
      __bf16 hb = (__bf16)f;
      h[t] = hb;
      l[t] = (__bf16)(f - (float)hb);
      s += f * f;
    }
    *(v4bf*)(ph + i) = h;
    *(v4bf*)(pl + i) = l;
  }
  #pragma unroll
  for (int off = 16; off > 0; off >>= 1) s += __shfl_xor(s, off, 32);
  if (lane == 0) xs[row] = s;
}

// ---------------------------------------------------------------------------
// pack W -> WThi/WTlo  TRANSPOSED [DOUT][DIN] bf16 (LDS-tiled 64x64 transpose)
// ---------------------------------------------------------------------------
#define WTP 72
__global__ __launch_bounds__(256) void pack_w_kernel(const float* __restrict__ W,
                                                     __bf16* __restrict__ WThi,
                                                     __bf16* __restrict__ WTlo) {
  __shared__ __bf16 sHi[64 * WTP];
  __shared__ __bf16 sLo[64 * WTP];
  const int c0 = blockIdx.x * 64;   // W column base
  const int k0 = blockIdx.y * 64;   // W row (K) base
  const int tid = threadIdx.x;

  #pragma unroll
  for (int i = 0; i < 4; ++i) {
    int idx = tid + i * 256;
    int r = idx >> 4, c4 = (idx & 15) * 4;
    v4f v = *(const v4f*)(W + (size_t)(k0 + r) * DOUT + c0 + c4);
    v4bf h, l;
    #pragma unroll
    for (int t = 0; t < 4; ++t) {
      float f = v[t];
      __bf16 hb = (__bf16)f;
      h[t] = hb;
      l[t] = (__bf16)(f - (float)hb);
    }
    *(v4bf*)(&sHi[r * WTP + c4]) = h;
    *(v4bf*)(&sLo[r * WTP + c4]) = l;
  }
  __syncthreads();

  int c = tid & 63;
  int q = tid >> 6;
  v16bf hv, lv;
  #pragma unroll
  for (int t = 0; t < 16; ++t) {
    hv[t] = sHi[(q * 16 + t) * WTP + c];
    lv[t] = sLo[(q * 16 + t) * WTP + c];
  }
  size_t o = (size_t)(c0 + c) * DIN + k0 + q * 16;
  *(v8u*)(WThi + o) = __builtin_bit_cast(v8u, hv);
  *(v8u*)(WTlo + o) = __builtin_bit_cast(v8u, lv);
}

// ---------------------------------------------------------------------------
// w_sq: two-stage (128-way parallel partials + reduce) to avoid latency bind
// ---------------------------------------------------------------------------
__global__ __launch_bounds__(256) void colsq_part_kernel(const float* __restrict__ W,
                                                         float* __restrict__ wsqp) {
  int col = blockIdx.x * 256 + threadIdx.x;
  int r0  = blockIdx.y * 128;
  float s = 0.f;
  #pragma unroll 16
  for (int i = 0; i < 128; ++i) {
    float v = W[(size_t)(r0 + i) * DOUT + col];
    s += v * v;
  }
  wsqp[(size_t)blockIdx.y * DOUT + col] = s;
}

__global__ __launch_bounds__(256) void colsq_reduce_kernel(const float* __restrict__ wsqp,
                                                           float* __restrict__ wsq) {
  int col = blockIdx.x * 256 + threadIdx.x;
  float s = 0.f;
  #pragma unroll
  for (int i = 0; i < 8; ++i) s += wsqp[(size_t)i * DOUT + col];
  wsq[col] = s;
}

// single-stage version for the small-workspace fallback path
__global__ __launch_bounds__(256) void colsq_kernel(const float* __restrict__ W,
                                                    float* __restrict__ wsq) {
  int col = blockIdx.x * 256 + threadIdx.x;
  float s = 0.f;
  #pragma unroll 8
  for (int i = 0; i < DIN; ++i) { float v = W[(size_t)i * DOUT + col]; s += v * v; }
  wsq[col] = s;
}

// ---------------------------------------------------------------------------
// main GEMM (pre-packed bf16): WG tile 128x128, wave tile 32x64, BK=32,
// double-buffered LDS filled by async global->LDS copies.
// out[b,o] = x.W - 0.5*(x_sq[b] + w_sq[o]) via hi*hi + hi*lo + lo*hi
// ---------------------------------------------------------------------------
#define BM 128
#define BN 128
#define BK 32
#define FP 40   // conflict-free LDS pitch (bf16); all chunks 16B aligned

__global__ __launch_bounds__(256) void gemm_bf16_kernel(const __bf16* __restrict__ Xhi,
                                                        const __bf16* __restrict__ Xlo,
                                                        const __bf16* __restrict__ WThi,
                                                        const __bf16* __restrict__ WTlo,
                                                        const float* __restrict__ xs,
                                                        const float* __restrict__ wsq,
                                                        float* __restrict__ out) {
  __shared__ __bf16 sXh[2][BM * FP], sXl[2][BM * FP];
  __shared__ __bf16 sWh[2][BN * FP], sWl[2][BN * FP];

  const int tid    = threadIdx.x;
  const int lane   = tid & 31;
  const int wave   = tid >> 5;
  const int wm     = wave & 3;    // 4 waves along M -> 32 rows each
  const int wn     = wave >> 2;   // 2 waves along N -> 64 cols each
  const int lane16 = lane & 15;
  const int half   = lane >> 4;

  const int row0 = blockIdx.y * BM;
  const int col0 = blockIdx.x * BN;

  v8f acc[2][4] = {};

  // issue global->LDS copies for one K-slab into buffer b
  auto stage = [&](int kk, int b) {
    #pragma unroll
    for (int i = 0; i < 2; ++i) {
      int idx = tid + i * 256;        // 512 16B chunks per array
      int r = idx >> 2, c8 = (idx & 3) * 8;
      size_t gx = (size_t)(row0 + r) * DIN + kk + c8;
      size_t gw = (size_t)(col0 + r) * DIN + kk + c8;
      int lo = r * FP + c8;
      cp16_g2l(Xhi  + gx, &sXh[b][lo]);
      cp16_g2l(Xlo  + gx, &sXl[b][lo]);
      cp16_g2l(WThi + gw, &sWh[b][lo]);
      cp16_g2l(WTlo + gw, &sWl[b][lo]);
    }
  };

  stage(0, 0);
  wait_async0();
  __syncthreads();

  for (int kk = 0; kk < DIN; kk += BK) {
    const int buf = (kk >> 5) & 1;
    if (kk + BK < DIN) stage(kk + BK, buf ^ 1);  // prefetch next slab (async)

    const __bf16* bXh = sXh[buf];
    const __bf16* bXl = sXl[buf];
    const __bf16* bWh = sWh[buf];
    const __bf16* bWl = sWl[buf];

    // A fragments: elems 0..7 -> K=half*8+t ; 8..15 -> K=half*8+16+t (gap 16)
    v16bf ah[2], al[2], bh[4], bl[4];
    #pragma unroll
    for (int ti = 0; ti < 2; ++ti) {
      int ro = (wm * 32 + ti * 16 + lane16) * FP + half * 8;
      ah[ti] = ldfrag(&bXh[ro], 16);
      al[ti] = ldfrag(&bXl[ro], 16);
    }
    // B fragments from transposed W: elems 0..7 -> K=half*8+t ; 8..15 -> K=half*8+16+t (gap 16)
    #pragma unroll
    for (int tj = 0; tj < 4; ++tj) {
      int ro = (wn * 64 + tj * 16 + lane16) * FP + half * 8;
      bh[tj] = ldfrag(&bWh[ro], 16);
      bl[tj] = ldfrag(&bWl[ro], 16);
    }

    #pragma unroll
    for (int ti = 0; ti < 2; ++ti)
      #pragma unroll
      for (int tj = 0; tj < 4; ++tj) {
        acc[ti][tj] = __builtin_amdgcn_wmma_f32_16x16x32_bf16(
            false, ah[ti], false, bh[tj], (short)0, acc[ti][tj], false, false);
        acc[ti][tj] = __builtin_amdgcn_wmma_f32_16x16x32_bf16(
            false, ah[ti], false, bl[tj], (short)0, acc[ti][tj], false, false);
        acc[ti][tj] = __builtin_amdgcn_wmma_f32_16x16x32_bf16(
            false, al[ti], false, bh[tj], (short)0, acc[ti][tj], false, false);
      }

    if (kk + BK < DIN) {
      wait_async0();       // own copies done; barrier makes all waves' visible
      __syncthreads();
    }
  }

  // epilogue: C/D layout: VGPR v -> M = v + 8*half ; N = lane16
  #pragma unroll
  for (int ti = 0; ti < 2; ++ti) {
    #pragma unroll
    for (int tj = 0; tj < 4; ++tj) {
      int col = col0 + wn * 64 + tj * 16 + lane16;
      float wv = wsq[col];
      #pragma unroll
      for (int v = 0; v < 8; ++v) {
        int row = row0 + wm * 32 + ti * 16 + half * 8 + v;
        out[(size_t)row * DOUT + col] = acc[ti][tj][v] - 0.5f * (xs[row] + wv);
      }
    }
  }
}

// ---------------------------------------------------------------------------
// Fallback path (small workspace): convert f32->bf16 split inside the GEMM
// ---------------------------------------------------------------------------
__global__ __launch_bounds__(256) void rowsq_kernel(const float* __restrict__ x,
                                                    float* __restrict__ xs) {
  int row  = blockIdx.x * 8 + (threadIdx.x >> 5);
  int lane = threadIdx.x & 31;
  const float* p = x + (size_t)row * DIN;
  float s = 0.f;
  #pragma unroll 4
  for (int i = lane; i < DIN; i += 32) { float v = p[i]; s += v * v; }
  #pragma unroll
  for (int off = 16; off > 0; off >>= 1) s += __shfl_xor(s, off, 32);
  if (lane == 0) xs[row] = s;
}

#define FBN 64
#define FXP 36
#define FWP 68
__global__ __launch_bounds__(256) void gemm_conv_kernel(const float* __restrict__ x,
                                                        const float* __restrict__ W,
                                                        const float* __restrict__ xs,
                                                        const float* __restrict__ wsq,
                                                        float* __restrict__ out) {
  __shared__ float sX[BM * FXP];
  __shared__ float sW[BK * FWP];
  const int tid = threadIdx.x, lane = tid & 31, wave = tid >> 5;
  const int wm = wave & 3, wn = wave >> 2;
  const int lane16 = lane & 15, half = lane >> 4;
  const int row0 = blockIdx.y * BM, col0 = blockIdx.x * FBN;

  v8f acc[2][2] = {};
  for (int kk = 0; kk < DIN; kk += BK) {
    #pragma unroll
    for (int i = 0; i < 4; ++i) {
      int idx = tid + i * 256;
      int r = idx >> 3, c4 = idx & 7;
      *(v4f*)(&sX[r * FXP + c4 * 4]) = *(const v4f*)(x + (size_t)(row0 + r) * DIN + kk + c4 * 4);
    }
    #pragma unroll
    for (int i = 0; i < 2; ++i) {
      int idx = tid + i * 256;
      int r = idx >> 4, c4 = idx & 15;
      *(v4f*)(&sW[r * FWP + c4 * 4]) = *(const v4f*)(W + (size_t)(kk + r) * DOUT + col0 + c4 * 4);
    }
    __syncthreads();

    v16bf ahi[2], alo[2], bhi[2], blo[2];
    #pragma unroll
    for (int ti = 0; ti < 2; ++ti) {
      const float* ap = &sX[(wm * 32 + ti * 16 + lane16) * FXP + half * 8];
      #pragma unroll
      for (int t = 0; t < 8; ++t) {
        splitf(ap[t],      t,     ahi[ti], alo[ti]);
        splitf(ap[16 + t], 8 + t, ahi[ti], alo[ti]);
      }
    }
    #pragma unroll
    for (int tj = 0; tj < 2; ++tj) {
      const float* bp = &sW[(half * 8) * FWP + wn * 32 + tj * 16 + lane16];
      #pragma unroll
      for (int t = 0; t < 16; ++t) splitf(bp[(t < 8 ? t : t + 8) * FWP], t, bhi[tj], blo[tj]);
    }
    #pragma unroll
    for (int ti = 0; ti < 2; ++ti)
      #pragma unroll
      for (int tj = 0; tj < 2; ++tj) {
        acc[ti][tj] = __builtin_amdgcn_wmma_f32_16x16x32_bf16(
            false, ahi[ti], false, bhi[tj], (short)0, acc[ti][tj], false, false);
        acc[ti][tj] = __builtin_amdgcn_wmma_f32_16x16x32_bf16(
            false, ahi[ti], false, blo[tj], (short)0, acc[ti][tj], false, false);
        acc[ti][tj] = __builtin_amdgcn_wmma_f32_16x16x32_bf16(
            false, alo[ti], false, bhi[tj], (short)0, acc[ti][tj], false, false);
      }
    __syncthreads();
  }
  #pragma unroll
  for (int ti = 0; ti < 2; ++ti)
    #pragma unroll
    for (int tj = 0; tj < 2; ++tj) {
      int col = col0 + wn * 32 + tj * 16 + lane16;
      float wv = wsq[col];
      #pragma unroll
      for (int v = 0; v < 8; ++v) {
        int row = row0 + wm * 32 + ti * 16 + half * 8 + v;
        out[(size_t)row * DOUT + col] = acc[ti][tj][v] - 0.5f * (xs[row] + wv);
      }
    }
}

// ---------------------------------------------------------------------------
extern "C" void kernel_launch(void* const* d_in, const int* in_sizes, int n_in,
                              void* d_out, int out_size, void* d_ws, size_t ws_size,
                              hipStream_t stream) {
  const float* x = (const float*)d_in[0];   // [2048,1024] f32
  const float* W = (const float*)d_in[1];   // [1024,4096] f32
  float* out = (float*)d_out;               // [2048,4096] f32

  char* wsb  = (char*)d_ws;
  float* xs  = (float*)wsb;                 // 2048 f32
  float* wsq = xs + BATCH;                  // 4096 f32
  float* wsqp = wsq + DOUT;                 // 8*4096 f32 partials (main path)

  const size_t packOff   = (size_t)256 * 1024;
  const size_t packBytes = packOff +
      2 * ((size_t)BATCH * DIN + (size_t)DOUT * DIN) * sizeof(__bf16);

  if (ws_size >= packBytes) {
    __bf16* Xhi  = (__bf16*)(wsb + packOff);
    __bf16* Xlo  = Xhi  + (size_t)BATCH * DIN;
    __bf16* WThi = Xlo  + (size_t)BATCH * DIN;
    __bf16* WTlo = WThi + (size_t)DOUT * DIN;

    pack_x_kernel<<<BATCH / 8, 256, 0, stream>>>(x, Xhi, Xlo, xs);
    pack_w_kernel<<<dim3(DOUT / 64, DIN / 64), 256, 0, stream>>>(W, WThi, WTlo);
    colsq_part_kernel<<<dim3(DOUT / 256, 8), 256, 0, stream>>>(W, wsqp);
    colsq_reduce_kernel<<<DOUT / 256, 256, 0, stream>>>(wsqp, wsq);
    dim3 grid(DOUT / BN, BATCH / BM);
    gemm_bf16_kernel<<<grid, 256, 0, stream>>>(Xhi, Xlo, WThi, WTlo, xs, wsq, out);
  } else {
    rowsq_kernel<<<BATCH / 8, 256, 0, stream>>>(x, xs);
    colsq_kernel<<<DOUT / 256, 256, 0, stream>>>(W, wsq);
    dim3 grid(DOUT / FBN, BATCH / BM);
    gemm_conv_kernel<<<grid, 256, 0, stream>>>(x, W, xs, wsq, out);
  }
}
